// GAT_53163105190183
// MI455X (gfx1250) — matrix operand order repaired and checked by reading, not compiled, not run on hardware
//
#include <hip/hip_runtime.h>
#include <hip/hip_bf16.h>

typedef __bf16 bf16;
typedef __attribute__((ext_vector_type(16))) __bf16 v16bf;
typedef __attribute__((ext_vector_type(8)))  __bf16 v8bf;
typedef __attribute__((ext_vector_type(8)))  float  v8f;

enum : int { Nn = 50000, Ee = 800000, Et = Ee + Nn, IN = 128, Hh = 4, Cc = 64, HC = Hh * Cc };

__device__ __forceinline__ int esrc(const int* ei, int e) { return e < Ee ? ei[e] : e - Ee; }
__device__ __forceinline__ int edst(const int* ei, int e) { return e < Ee ? ei[Ee + e] : e - Ee; }

__device__ __forceinline__ unsigned fkey(float f) {
  unsigned u = __float_as_uint(f);
  return (u & 0x80000000u) ? ~u : (u | 0x80000000u);
}
__device__ __forceinline__ float fdecode(unsigned k) {
  return (k & 0x80000000u) ? __uint_as_float(k & 0x7fffffffu) : __uint_as_float(~k);
}

// ---------------- setup kernels ----------------

__global__ void zero_colsum(float* colsum) { colsum[threadIdx.x] = 0.f; }

__global__ void colmean_acc(const float* __restrict__ ea, float* __restrict__ colsum) {
  int c = threadIdx.x;               // 128 threads = one per column
  float acc = 0.f;
  for (int e = blockIdx.x; e < Ee; e += gridDim.x) acc += ea[(size_t)e * IN + c];
  atomicAdd(&colsum[c], acc);
}

__global__ void colmean_fin(const float* colsum, float* ea_mean) {
  ea_mean[threadIdx.x] = colsum[threadIdx.x] * (1.0f / (float)Ee);
}

// v[l][k][h] = sum_c We[l][h*64+c][k] * att_edge[l][h][c]
__global__ void foldv(const float* __restrict__ We, const float* __restrict__ atts,
                      float* __restrict__ v) {
  int l = blockIdx.x;
  int tid = threadIdx.x;             // 512 = 128 k * 4 h
  int k = tid >> 2, h = tid & 3;
  const float* Wl = We + (size_t)l * HC * IN;
  const float* ae = atts + (size_t)l * 3 * HC + 2 * HC + h * Cc;
  float acc = 0.f;
  for (int c = 0; c < Cc; ++c) acc += Wl[(size_t)(h * Cc + c) * IN + k] * ae[c];
  v[((size_t)l * IN + k) * 4 + h] = acc;
}

__global__ void f32_to_bf16(const float* __restrict__ in, bf16* __restrict__ out, int n) {
  for (int i = blockIdx.x * blockDim.x + threadIdx.x; i < n; i += gridDim.x * blockDim.x)
    out[i] = (bf16)in[i];
}

// ---------------- per-layer kernels ----------------

__global__ void layer_init(float* __restrict__ agg, unsigned* __restrict__ amaxkey,
                           float* __restrict__ denom) {
  int i = blockIdx.x * blockDim.x + threadIdx.x;
  if (i < Nn * HC) agg[i] = 0.f;
  if (i < Nn * Hh) { amaxkey[i] = 0u; denom[i] = 0.f; }
}

// C[M,256] = A[M,K] (bf16) x B^T, B = W [256,K] (bf16 row-major).
// One wave per 16x64 output tile; 4 WMMAs per k-step.
__global__ __launch_bounds__(128)
void gemm_bf16_wmma(const bf16* __restrict__ A, const bf16* __restrict__ B,
                    float* __restrict__ Cmat, int K) {
  const int lane = threadIdx.x & 31;
  const int wave = threadIdx.x >> 5;       // 0..3 -> 64-col group
  const int m0   = blockIdx.x * 16;
  const int col0 = wave * 64;
  const int half = lane >> 4;              // lane group 0/1
  const int l16  = lane & 15;

  v8f acc[4] = {};
  const bf16* arow = A + (size_t)(m0 + l16) * K;

  for (int k0 = 0; k0 < K; k0 += 32) {
    // A fragment: per lane 16 contiguous K={half*16..+15} (matches B)
    v8bf alo = *(const v8bf*)(arow + k0 + half * 16);
    v8bf ahi = *(const v8bf*)(arow + k0 + half * 16 + 8);
    v16bf af;
#pragma unroll
    for (int i = 0; i < 8; ++i) { af[i] = alo[i]; af[i + 8] = ahi[i]; }

    if (k0 + 32 < K) __builtin_prefetch(arow + k0 + 32, 0, 0);

#pragma unroll
    for (int t = 0; t < 4; ++t) {
      // B fragment: column n = col0+t*16+l16, 16 contiguous k starting at 16*half
      const bf16* bp = B + (size_t)(col0 + t * 16 + l16) * K + k0 + half * 16;
      v16bf bfrag = *(const v16bf*)bp;
      acc[t] = __builtin_amdgcn_wmma_f32_16x16x32_bf16(
          false, af, false, bfrag, (short)0, acc[t], false, false);
    }
  }

#pragma unroll
  for (int t = 0; t < 4; ++t) {
    float* cp = Cmat + (size_t)(m0 + half * 8) * HC + col0 + t * 16 + l16;
#pragma unroll
    for (int r = 0; r < 8; ++r) cp[(size_t)r * HC] = acc[t][r];
  }
}

// per (n,h): a_src / a_dst dots with attention vectors
__global__ void asrc_adst(const float* __restrict__ xp, const float* __restrict__ atts,
                          float* __restrict__ asrc, float* __restrict__ adst, int layer) {
  int i = blockIdx.x * blockDim.x + threadIdx.x;
  if (i >= Nn * Hh) return;
  int n = i >> 2, h = i & 3;
  const float* xr = xp + (size_t)n * HC + h * Cc;
  const float* a0 = atts + (size_t)layer * 3 * HC + h * Cc;
  const float* a1 = a0 + HC;
  float s0 = 0.f, s1 = 0.f;
  for (int c = 0; c < Cc; ++c) { float x = xr[c]; s0 += x * a0[c]; s1 += x * a1[c]; }
  asrc[i] = s0; adst[i] = s1;
}

// one wave per edge: a_edge = ea_row . v_l (folded), then logit + leakyrelu + atomic max
__global__ void edge_attn(const int* __restrict__ ei, const float* __restrict__ ea,
                          const float* __restrict__ ea_mean, const float* __restrict__ v,
                          const float* __restrict__ asrc, const float* __restrict__ adst,
                          float* __restrict__ a, unsigned* __restrict__ amaxkey, int layer) {
  int gid = blockIdx.x * blockDim.x + threadIdx.x;
  int e = gid >> 5, lane = gid & 31;
  if (e >= Et) return;
  const float* row = (e < Ee) ? (ea + (size_t)e * IN) : ea_mean;
  float4 r4 = *(const float4*)(row + lane * 4);
  const float* vb = v + (size_t)layer * IN * 4;
  float rv[4] = {r4.x, r4.y, r4.z, r4.w};
  float p0 = 0.f, p1 = 0.f, p2 = 0.f, p3 = 0.f;
#pragma unroll
  for (int q = 0; q < 4; ++q) {
    float4 vv = *(const float4*)(vb + (size_t)(lane * 4 + q) * 4);
    p0 += rv[q] * vv.x; p1 += rv[q] * vv.y; p2 += rv[q] * vv.z; p3 += rv[q] * vv.w;
  }
#pragma unroll
  for (int m = 16; m > 0; m >>= 1) {
    p0 += __shfl_xor(p0, m); p1 += __shfl_xor(p1, m);
    p2 += __shfl_xor(p2, m); p3 += __shfl_xor(p3, m);
  }
  int s = esrc(ei, e), d = edst(ei, e);
  if (lane < 4) {
    float pe = (lane == 0) ? p0 : (lane == 1) ? p1 : (lane == 2) ? p2 : p3;
    float val = asrc[s * 4 + lane] + adst[d * 4 + lane] + pe;
    val = val > 0.f ? val : 0.2f * val;      // leaky_relu(0.2)
    a[(size_t)e * 4 + lane] = val;
    atomicMax(amaxkey + (size_t)d * 4 + lane, fkey(val));
  }
}

__global__ void edge_exp(const int* __restrict__ ei, float* __restrict__ a,
                         const unsigned* __restrict__ amaxkey, float* __restrict__ denom) {
  int i = blockIdx.x * blockDim.x + threadIdx.x;
  if (i >= Et * Hh) return;
  int e = i >> 2, h = i & 3;
  int d = edst(ei, e);
  float m = fdecode(amaxkey[d * 4 + h]);
  float ex = __expf(a[i] - m);
  a[i] = ex;
  atomicAdd(&denom[d * 4 + h], ex);
}

__global__ void edge_norm(const int* __restrict__ ei, float* __restrict__ a,
                          const float* __restrict__ denom) {
  int i = blockIdx.x * blockDim.x + threadIdx.x;
  if (i >= Et * Hh) return;
  int e = i >> 2, h = i & 3;
  a[i] = a[i] / denom[edst(ei, e) * 4 + h];
}

// scatter-add: agg[dst] += a[e,h] * xp[src]; one thread = 4 consecutive columns
__global__ void aggregate(const int* __restrict__ ei, const float* __restrict__ a,
                          const float* __restrict__ xp, float* __restrict__ agg) {
  int i = blockIdx.x * blockDim.x + threadIdx.x;
  if (i >= Et * 64) return;
  int e = i >> 6, q = i & 63;
  int s = esrc(ei, e), d = edst(ei, e);
  float w = a[(size_t)e * 4 + (q >> 4)];
  float4 xv = ((const float4*)xp)[(size_t)s * 64 + q];
  float* o = agg + (size_t)d * HC + q * 4;
  atomicAdd(o + 0, w * xv.x); atomicAdd(o + 1, w * xv.y);
  atomicAdd(o + 2, w * xv.z); atomicAdd(o + 3, w * xv.w);
}

// head-mean + bias (+BN+ReLU for layers 0,1); emit bf16 activations or final f32
__global__ void finish(const float* __restrict__ agg, const float* __restrict__ biases,
                       const float* __restrict__ bg, const float* __restrict__ bb,
                       const float* __restrict__ bm, const float* __restrict__ bv,
                       bf16* __restrict__ hbf, float* __restrict__ out, int layer) {
  int i = blockIdx.x * blockDim.x + threadIdx.x;
  if (i >= Nn * Cc) return;
  int n = i >> 6, c = i & 63;
  const float* g = agg + (size_t)n * HC + c;
  float s = (g[0] + g[64] + g[128] + g[192]) * 0.25f + biases[layer * Cc + c];
  if (layer < 2) {
    s = (s - bm[layer * Cc + c]) * rsqrtf(bv[layer * Cc + c] + 1e-5f) * bg[layer * Cc + c]
        + bb[layer * Cc + c];
    s = s > 0.f ? s : 0.f;
    hbf[i] = (bf16)s;
  } else {
    out[i] = s;
  }
}

// ---------------- host launch ----------------

extern "C" void kernel_launch(void* const* d_in, const int* in_sizes, int n_in,
                              void* d_out, int out_size, void* d_ws, size_t ws_size,
                              hipStream_t stream) {
  const float* x    = (const float*)d_in[0];
  const int*   ei   = (const int*)d_in[1];
  const float* ea   = (const float*)d_in[2];
  const float* W0   = (const float*)d_in[3];
  const float* W12  = (const float*)d_in[4];
  const float* We   = (const float*)d_in[5];
  const float* atts = (const float*)d_in[6];
  const float* bias = (const float*)d_in[7];
  const float* bg   = (const float*)d_in[8];
  const float* bb   = (const float*)d_in[9];
  const float* bm   = (const float*)d_in[10];
  const float* bv   = (const float*)d_in[11];
  float* out = (float*)d_out;

  char* ws = (char*)d_ws;
  size_t off = 0;
  auto alloc = [&](size_t bytes) { void* p = ws + off; off += (bytes + 255) & ~(size_t)255; return p; };

  float*    colsum  = (float*)alloc(IN * 4);
  float*    ea_mean = (float*)alloc(IN * 4);
  float*    v       = (float*)alloc(3 * IN * 4 * 4);
  bf16*     W0bf    = (bf16*)alloc((size_t)HC * IN * 2);
  bf16*     W12bf   = (bf16*)alloc((size_t)2 * HC * Cc * 2);
  bf16*     hbf     = (bf16*)alloc((size_t)Nn * IN * 2);
  float*    xp      = (float*)alloc((size_t)Nn * HC * 4);
  float*    agg     = (float*)alloc((size_t)Nn * HC * 4);
  float*    asrcb   = (float*)alloc((size_t)Nn * Hh * 4);
  float*    adstb   = (float*)alloc((size_t)Nn * Hh * 4);
  unsigned* amaxkey = (unsigned*)alloc((size_t)Nn * Hh * 4);
  float*    denom   = (float*)alloc((size_t)Nn * Hh * 4);
  float*    abuf    = (float*)alloc((size_t)Et * Hh * 4);
  (void)ws_size; (void)in_sizes; (void)n_in; (void)out_size;

  // edge_attr passthrough (tuple output #2)
  hipMemcpyAsync(out + (size_t)Nn * Cc, ea, (size_t)Ee * IN * sizeof(float),
                 hipMemcpyDeviceToDevice, stream);

  // setup
  zero_colsum<<<1, 128, 0, stream>>>(colsum);
  colmean_acc<<<1024, 128, 0, stream>>>(ea, colsum);
  colmean_fin<<<1, 128, 0, stream>>>(colsum, ea_mean);
  foldv<<<3, 512, 0, stream>>>(We, atts, v);
  f32_to_bf16<<<64, 256, 0, stream>>>(W0, W0bf, HC * IN);
  f32_to_bf16<<<64, 256, 0, stream>>>(W12, W12bf, 2 * HC * Cc);
  f32_to_bf16<<<2048, 256, 0, stream>>>(x, hbf, Nn * IN);

  for (int l = 0; l < 3; ++l) {
    const int K = (l == 0) ? IN : Cc;
    const bf16* B = (l == 0) ? W0bf : (W12bf + (size_t)(l - 1) * HC * Cc);

    layer_init<<<(Nn * HC + 255) / 256, 256, 0, stream>>>(agg, amaxkey, denom);
    gemm_bf16_wmma<<<Nn / 16, 128, 0, stream>>>(hbf, B, xp, K);
    asrc_adst<<<(Nn * Hh + 255) / 256, 256, 0, stream>>>(xp, atts, asrcb, adstb, l);
    edge_attn<<<(Et * 32 + 255) / 256, 256, 0, stream>>>(ei, ea, ea_mean, v, asrcb, adstb,
                                                         abuf, amaxkey, l);
    edge_exp<<<(Et * Hh + 255) / 256, 256, 0, stream>>>(ei, abuf, amaxkey, denom);
    edge_norm<<<(Et * Hh + 255) / 256, 256, 0, stream>>>(ei, abuf, denom);
    aggregate<<<(Et * 64 + 255) / 256, 256, 0, stream>>>(ei, abuf, xp, agg);
    finish<<<(Nn * Cc + 255) / 256, 256, 0, stream>>>(agg, bias, bg, bb, bm, bv, hbf, out, l);
  }
}
